// MultiHeadAttention_16801912062315
// MI455X (gfx1250) — matrix operand order repaired and checked by reading, not compiled, not run on hardware
//
#include <hip/hip_runtime.h>
#include <hip/hip_bf16.h>

// ---------------------------------------------------------------------------
// MI455X / gfx1250 multi-head attention, bf16 WMMA path (f32 accumulate).
// B=32, S=64, H=8, E=1024. Compute-bound: 137 GFLOP of GEMM vs ~5.8us of HBM,
// so bf16 matrix cores (K=32/instr) over f32 WMMA (K=4/instr) is the win.
// ---------------------------------------------------------------------------

typedef __attribute__((ext_vector_type(16))) __bf16 v16bf;
typedef __attribute__((ext_vector_type(8)))  float  v8f;

union FragB16 {
    v16bf v;
    uint4 q[2];
};

__device__ __forceinline__ __bf16 f2bf(float f) { return (__bf16)f; }

// ---------------------------------------------------------------------------
// C = A * B^T GEMM.  A: MxK (f32 or bf16, row-major), B: NxK f32 row-major.
// Converts to bf16 while staging to LDS.  Block tile 128x128, K-step 32,
// 8 waves each computing 32x64 (2x4 WMMA tiles).  Software-pipelined:
// next K-tile global loads issue before the current tile's WMMAs.
//   MODE 0: out bf16, layout (B,H,S,E), multiplier mask[b,e]*scale  (Q,K)
//   MODE 1: out bf16, layout (B,H,E,S) transposed, multiplier mask  (V^T)
//   MODE 2: out f32 d_out, (acc + bias[n]) * mask[b,n]              (final)
// ---------------------------------------------------------------------------
#define BM 128
#define BN 128
#define BKSTEP 32
#define APITCH 40   // bf16 pitch: 80 bytes/row -> 16B-aligned rows, spreads banks

template<int MODE, bool ABF16>
__global__ __launch_bounds__(256)
void gemm_bf16_wmma(const void* __restrict__ Av,
                    const float* __restrict__ Bw,
                    void* __restrict__ Cout,
                    const float* __restrict__ mask,
                    const float* __restrict__ bias,
                    int M, int N, int K, float scale)
{
    __shared__ alignas(16) __bf16 ldsA[BM * APITCH];
    __shared__ alignas(16) __bf16 ldsB[BN * APITCH];

    const int tid  = threadIdx.x;
    const int w    = tid >> 5;
    const int lane = tid & 31;
    const int lm   = lane & 15;   // M (A) / N (B) index within 16
    const int kh   = lane >> 4;   // K-half selector
    const int wm   = (w & 3) * 32;
    const int wn   = (w >> 2) * 64;
    const int bm   = blockIdx.y * BM;
    const int bn   = blockIdx.x * BN;

    const float*  Af = (const float*)Av;
    const __bf16* Ab = (const __bf16*)Av;

    // register staging buffers (software pipeline stage)
    float4 raf[4];
    uint4  rab[2];
    float4 rb[4];

    auto loadTile = [&](int kk) {
        if (ABF16) {
#pragma unroll
            for (int i = 0; i < 2; ++i) {
                int idx = tid + i * 256;              // 512 x uint4 (8 bf16)
                int row = idx >> 2, c8 = idx & 3;
                rab[i] = *(const uint4*)(Ab + (size_t)(bm + row) * K + kk + c8 * 8);
            }
        } else {
#pragma unroll
            for (int i = 0; i < 4; ++i) {
                int idx = tid + i * 256;              // 1024 x float4
                int row = idx >> 3, c4 = idx & 7;
                raf[i] = *(const float4*)(Af + (size_t)(bm + row) * K + kk + c4 * 4);
            }
        }
#pragma unroll
        for (int i = 0; i < 4; ++i) {
            int idx = tid + i * 256;
            int row = idx >> 3, c4 = idx & 7;
            rb[i] = *(const float4*)(Bw + (size_t)(bn + row) * K + kk + c4 * 4);
        }
    };

    auto storeTile = [&]() {
        if (ABF16) {
#pragma unroll
            for (int i = 0; i < 2; ++i) {
                int idx = tid + i * 256;
                int row = idx >> 2, c8 = idx & 3;
                *(uint4*)&ldsA[row * APITCH + c8 * 8] = rab[i];
            }
        } else {
#pragma unroll
            for (int i = 0; i < 4; ++i) {
                int idx = tid + i * 256;
                int row = idx >> 3, c4 = idx & 7;
                __bf16* p = &ldsA[row * APITCH + c4 * 4];
                p[0] = f2bf(raf[i].x); p[1] = f2bf(raf[i].y);
                p[2] = f2bf(raf[i].z); p[3] = f2bf(raf[i].w);
            }
        }
#pragma unroll
        for (int i = 0; i < 4; ++i) {
            int idx = tid + i * 256;
            int row = idx >> 3, c4 = idx & 7;
            __bf16* p = &ldsB[row * APITCH + c4 * 4];
            p[0] = f2bf(rb[i].x); p[1] = f2bf(rb[i].y);
            p[2] = f2bf(rb[i].z); p[3] = f2bf(rb[i].w);
        }
    };

    v8f acc[2][4] = {};
    const int nk = K / BKSTEP;
    loadTile(0);

    for (int kt = 0; kt < nk; ++kt) {
        storeTile();
        __syncthreads();

        // issue next tile's global loads NOW so HBM latency hides behind WMMAs
        if (kt + 1 < nk) loadTile((kt + 1) * BKSTEP);
        // L2 prefetch two tiles ahead on the weight stream (dominant traffic)
        if (kt + 2 < nk) {
#pragma unroll
            for (int i = 0; i < 4; ++i) {
                int idx = tid + i * 256;
                int row = idx >> 3, c4 = idx & 7;
                __builtin_prefetch(Bw + (size_t)(bn + row) * K + (kt + 2) * BKSTEP + c4 * 4, 0, 1);
            }
        }

        // ---- load fragments per ISA VGPR layout, run 8 WMMAs ----
        FragB16 a[2], b[4];
#pragma unroll
        for (int mi = 0; mi < 2; ++mi) {
            // lanes 0-15: K=0..7 & 16..23 ; lanes 16-31: K=8..15 & 24..31
            const __bf16* base = &ldsA[(wm + mi * 16 + lm) * APITCH + kh * 8];
            a[mi].q[0] = *(const uint4*)(base);
            a[mi].q[1] = *(const uint4*)(base + 16);
        }
#pragma unroll
        for (int ni = 0; ni < 4; ++ni) {
            // lanes 0-15: K=0..7 & 16..23 ; lanes 16-31: K=8..15 & 24..31
            const __bf16* base = &ldsB[(wn + ni * 16 + lm) * APITCH + kh * 8];
            b[ni].q[0] = *(const uint4*)(base);
            b[ni].q[1] = *(const uint4*)(base + 16);
        }
#pragma unroll
        for (int mi = 0; mi < 2; ++mi)
#pragma unroll
            for (int ni = 0; ni < 4; ++ni)
                acc[mi][ni] = __builtin_amdgcn_wmma_f32_16x16x32_bf16(
                    false, a[mi].v, false, b[ni].v, (short)0, acc[mi][ni],
                    false, false);

        __syncthreads();
    }

    // ---- epilogue: C/D layout -> lane l, vgpr r: M = 8*(l>>4)+r, N = l&15 ----
    const int E = 1024;
#pragma unroll
    for (int mi = 0; mi < 2; ++mi) {
#pragma unroll
        for (int ni = 0; ni < 4; ++ni) {
#pragma unroll
            for (int r = 0; r < 8; ++r) {
                int m = bm + wm + mi * 16 + 8 * kh + r;
                int n = bn + wn + ni * 16 + lm;
                float vacc = acc[mi][ni][r];
                if (MODE == 2) {
                    float o = (vacc + bias[n]) * mask[(m >> 6) * E + n];
                    ((float*)Cout)[(size_t)m * N + n] = o;
                } else {
                    int bb = m >> 6, s = m & 63;
                    int h  = n >> 10, e = n & 1023;
                    float mul = mask[bb * E + e] * scale;
                    __bf16 o = f2bf(vacc * mul);
                    if (MODE == 0)  // (B,H,S,E)
                        ((__bf16*)Cout)[((size_t)(bb * 8 + h) * 64 + s) * 1024 + e] = o;
                    else            // (B,H,E,S) transposed (V^T for A@V B-frags)
                        ((__bf16*)Cout)[((size_t)(bb * 8 + h) * 1024 + e) * 64 + s] = o;
                }
            }
        }
    }
}

// ---------------------------------------------------------------------------
// Per-(b,h) attention: scores = q k^T (64x64, K=1024 bf16 WMMA, f32 acc),
// f32 softmax in LDS, P -> bf16, y = P v via WMMA (K=64), write y (B,S,H*E).
// One workgroup (8 waves) per (b,h); q/k/vT fragments loaded straight from
// global (hot in L2: 384 KB per (b,h)).
// ---------------------------------------------------------------------------
__global__ __launch_bounds__(256)
void attention_wmma(const __bf16* __restrict__ qg,
                    const __bf16* __restrict__ kg,
                    const __bf16* __restrict__ vTg,
                    __bf16* __restrict__ yg)
{
    __shared__ alignas(16) float  sc[64 * 65];
    __shared__ alignas(16) __bf16 pb[64 * 72];   // 144B rows: 16B-aligned

    const int bh   = blockIdx.x;          // b*8 + h
    const int tid  = threadIdx.x;
    const int w    = tid >> 5;
    const int lane = tid & 31;
    const int lm   = lane & 15;
    const int kh   = lane >> 4;

    const __bf16* qb = qg  + (size_t)bh * 64 * 1024;
    const __bf16* kb = kg  + (size_t)bh * 64 * 1024;
    const __bf16* vb = vTg + (size_t)bh * 1024 * 64;

    // ---- phase A: scores (4x4 grid of 16x16 tiles; 2 tiles per wave) ----
#pragma unroll
    for (int t = 0; t < 2; ++t) {
        const int tile = w * 2 + t;
        const int ti = tile >> 2, tj = tile & 3;
        v8f acc = {};
        for (int kk = 0; kk < 1024; kk += 32) {
            FragB16 a, b;
            const __bf16* ab = qb + (size_t)(ti * 16 + lm) * 1024 + kk + kh * 8;
            a.q[0] = *(const uint4*)(ab);
            a.q[1] = *(const uint4*)(ab + 16);
            const __bf16* bb = kb + (size_t)(tj * 16 + lm) * 1024 + kk + kh * 8;
            b.q[0] = *(const uint4*)(bb);
            b.q[1] = *(const uint4*)(bb + 16);
            acc = __builtin_amdgcn_wmma_f32_16x16x32_bf16(
                false, a.v, false, b.v, (short)0, acc, false, false);
        }
#pragma unroll
        for (int r = 0; r < 8; ++r)
            sc[(ti * 16 + 8 * kh + r) * 65 + tj * 16 + lm] = acc[r];
    }
    __syncthreads();

    // ---- softmax over rows (one row per thread, 64 active) ----
    if (tid < 64) {
        float mx = -3.4e38f;
        for (int j = 0; j < 64; ++j) mx = fmaxf(mx, sc[tid * 65 + j]);
        float sum = 0.f;
        for (int j = 0; j < 64; ++j) {
            float e = __expf(sc[tid * 65 + j] - mx);
            sc[tid * 65 + j] = e;
            sum += e;
        }
        float inv = 1.0f / sum;
        for (int j = 0; j < 64; ++j)
            pb[tid * 72 + j] = f2bf(sc[tid * 65 + j] * inv);
    }
    __syncthreads();

    // ---- phase B: y = P @ v.  wave -> m-tile (w>>1), n-half (w&1) ----
    const int mi   = w >> 1;
    const int half = w & 1;
    FragB16 pa[2];
#pragma unroll
    for (int ks = 0; ks < 2; ++ks) {
        const __bf16* base = &pb[(mi * 16 + lm) * 72 + ks * 32 + kh * 8];
        pa[ks].q[0] = *(const uint4*)(base);
        pa[ks].q[1] = *(const uint4*)(base + 16);
    }
    const int b_ = bh >> 3, h_ = bh & 7;
    for (int nt = 0; nt < 32; ++nt) {
        const int n0 = half * 512 + nt * 16;
        v8f acc = {};
#pragma unroll
        for (int ks = 0; ks < 2; ++ks) {
            FragB16 bf;
            const __bf16* base = vb + (size_t)(n0 + lm) * 64 + ks * 32 + kh * 8;
            bf.q[0] = *(const uint4*)(base);
            bf.q[1] = *(const uint4*)(base + 16);
            acc = __builtin_amdgcn_wmma_f32_16x16x32_bf16(
                false, pa[ks].v, false, bf.v, (short)0, acc, false, false);
        }
#pragma unroll
        for (int r = 0; r < 8; ++r) {
            int s = mi * 16 + 8 * kh + r;
            int e = n0 + lm;
            yg[((size_t)(b_ * 64 + s) * 8192) + h_ * 1024 + e] = f2bf(acc[r]);
        }
    }
}

// ---------------------------------------------------------------------------
extern "C" void kernel_launch(void* const* d_in, const int* in_sizes, int n_in,
                              void* d_out, int out_size, void* d_ws, size_t ws_size,
                              hipStream_t stream)
{
    (void)in_sizes; (void)n_in; (void)out_size; (void)ws_size;

    const float* x    = (const float*)d_in[0];   // (32,64,1024)
    const float* mask = (const float*)d_in[1];   // (32,1024)
    const float* Wq   = (const float*)d_in[2];   // (8192,1024)
    const float* Wk   = (const float*)d_in[3];
    const float* Wv   = (const float*)d_in[4];
    const float* Wo   = (const float*)d_in[5];   // (1024,8192)
    const float* bo   = (const float*)d_in[6];   // (1024)

    const int Bsz = 32, Hh = 8, Ee = 1024;
    const int M   = Bsz * 64;     // 2048
    const int NQK = Hh * Ee;      // 8192

    // workspace: q 32MB | k 32MB | vT 32MB | y 32MB  (128 MB total, bf16)
    char* ws = (char*)d_ws;
    __bf16* qg  = (__bf16*)(ws);
    __bf16* kg  = (__bf16*)(ws + (size_t)32 * 1024 * 1024);
    __bf16* vTg = (__bf16*)(ws + (size_t)64 * 1024 * 1024);
    __bf16* yg  = (__bf16*)(ws + (size_t)96 * 1024 * 1024);

    dim3 blk(256);
    dim3 gqkv(NQK / BN, M / BM);          // 64 x 16 workgroups
    const float scale = 1.0f / 32.0f;     // E^-0.5

    gemm_bf16_wmma<0, false><<<gqkv, blk, 0, stream>>>(x, Wq, qg,  mask, nullptr, M, NQK, Ee, scale);
    gemm_bf16_wmma<0, false><<<gqkv, blk, 0, stream>>>(x, Wk, kg,  mask, nullptr, M, NQK, Ee, 1.0f);
    gemm_bf16_wmma<1, false><<<gqkv, blk, 0, stream>>>(x, Wv, vTg, mask, nullptr, M, NQK, Ee, 1.0f);

    attention_wmma<<<dim3(Bsz * Hh), blk, 0, stream>>>(qg, kg, vTg, yg);

    dim3 gout(Ee / BN, M / BM);           // 8 x 16
    gemm_bf16_wmma<2, true><<<gout, blk, 0, stream>>>(yg, Wo, d_out, mask, bo, M, Ee, NQK, 1.0f);
}
